// TransformerEncoderLayerWithMoE_60928406061077
// MI455X (gfx1250) — compile-verified, matrix operand order checked by reading, not run on hardware
//
#include <hip/hip_runtime.h>
#include <hip/hip_bf16.h>
#include <stdint.h>

// ---------------------------------------------------------------------------
// TransformerEncoderLayer + top-2 MoE for MI455X (gfx1250, wave32, WMMA).
// GEMMs: v_wmma_f32_16x16x32_bf16, double-buffered LDS tiles filled with
// global_load_async_to_lds_b128 (ASYNCcnt), 64x64 wave tiles (1:1 ds:wmma).
// ---------------------------------------------------------------------------

typedef __attribute__((ext_vector_type(16))) __bf16 v16bf;
typedef __attribute__((ext_vector_type(8)))  __bf16 v8bf;
typedef __attribute__((ext_vector_type(8)))  float  v8f;

#define BB   8
#define SS   1024
#define DD   1024
#define HHH  16
#define DHH  64
#define EE   8
#define HIDD 4096
#define TT   (BB * SS)   // 8192 tokens

union FragU { v16bf v; v8bf h[2]; };
union AccU  { v8f v; float f[8]; };

__device__ __forceinline__ v8f vzero8f() {
  v8f z = {0.f, 0.f, 0.f, 0.f, 0.f, 0.f, 0.f, 0.f};
  return z;
}

// Load one 16x32 bf16 A/B fragment from an LDS tile (row-major, stride `ld`
// elems, 16B-aligned rows). Lane l: row = row0 + (l&15); K pattern per ISA:
// elems 0..7 -> K = col0 + 8g + i, elems 8..15 -> K = col0 + 16 + 8g + (i-8).
__device__ __forceinline__ v16bf ld_frag(const __bf16* t, int ld, int row0,
                                         int col0, int lane) {
  const int g = lane >> 4, r = lane & 15;
  const __bf16* p = t + (row0 + r) * ld + col0 + g * 8;
  FragU f;
  f.h[0] = *(const v8bf*)(p);
  f.h[1] = *(const v8bf*)(p + 16);
  return f.v;
}

__device__ __forceinline__ v8f wmma_bf16f32(v16bf a, v16bf b, v8f c) {
  return __builtin_amdgcn_wmma_f32_16x16x32_bf16(false, a, false, b, (short)0,
                                                 c, false, false);
}

// Async copy 16B global -> LDS (CDNA5 TDM-lite path, tracked by ASYNCcnt).
__device__ __forceinline__ void async_cp16(uint32_t lds_addr, const void* gptr) {
  const unsigned long long ga = (unsigned long long)(uintptr_t)gptr;
  asm volatile("global_load_async_to_lds_b128 %0, %1, off"
               :: "v"(lds_addr), "v"(ga) : "memory");
}
__device__ __forceinline__ void wait_async0() {
  asm volatile("s_wait_asynccnt 0" ::: "memory");
}

// ---------------------------------------------------------------------------
// WMMA GEMM: C[M,N] = A[M,K] * B[N,K]^T + bias[N]   (A, B bf16)
//   OMODE  : 0 = bf16 out, 1 = f32 out, 2 = scatter-add f32 (row_map+scale)
//   RELU   : relu after bias
//   GATHER : A rows gathered through row_map (clamped; dead rows guarded)
// Block 128 thr (4 waves), tile 128x128x32, wave tile 64x64, double-buffered
// LDS filled with async loads; next tile fetched while current computes.
// ---------------------------------------------------------------------------
template <int OMODE, bool RELU, bool GATHER>
__global__ __launch_bounds__(128) void gemm_wmma(
    const __bf16* __restrict__ A, const __bf16* __restrict__ Bw,
    const float* __restrict__ bias, int M, int N, int K,
    __bf16* __restrict__ out_bf16, float* __restrict__ out_f32,
    const int* __restrict__ row_map, const int* __restrict__ cnt_ptr,
    const float* __restrict__ scale, float* __restrict__ scat, int ldscat) {
  constexpr int BM = 128, BN = 128, BK = 32, LDT = 40;
  __shared__ __bf16 As[2][BM * LDT];
  __shared__ __bf16 Bs[2][BN * LDT];

  const int Meff = cnt_ptr ? *cnt_ptr : M;
  const int m0 = blockIdx.y * BM;
  if (m0 >= Meff) return;
  const int n0 = blockIdx.x * BN;
  const int tid = threadIdx.x;
  const int lane = tid & 31, wv = tid >> 5;
  const int wm0 = (wv >> 1) * 64, wn0 = (wv & 1) * 64;

  // Staging: 128 rows x 32 cols per tile, 16B chunks; 4 A + 4 B chunks/thread.
  long arow_off[4];  // element offset of A row start (+colchunk), k0 added later
#pragma unroll
  for (int it = 0; it < 4; it++) {
    const int c = tid + it * 128;
    const int row = c >> 2, cc = c & 3;
    long srow;
    if constexpr (GATHER) {
      const int gr = m0 + row;
      const int gi = gr < Meff ? gr : (Meff - 1);  // keep address in-bounds
      srow = row_map[gi];
    } else {
      srow = m0 + row;
    }
    arow_off[it] = srow * (long)K + cc * 8;
  }

  v8f acc[4][4];
#pragma unroll
  for (int i = 0; i < 4; i++)
#pragma unroll
    for (int j = 0; j < 4; j++) acc[i][j] = vzero8f();

  auto issue = [&](int buf, int k0) {
#pragma unroll
    for (int it = 0; it < 4; it++) {
      const int c = tid + it * 128;
      const int row = c >> 2, cc = c & 3;
      const uint32_t la = (uint32_t)(uintptr_t)&As[buf][row * LDT + cc * 8];
      async_cp16(la, A + arow_off[it] + k0);
      const uint32_t lb = (uint32_t)(uintptr_t)&Bs[buf][row * LDT + cc * 8];
      async_cp16(lb, Bw + (long)(n0 + row) * K + k0 + cc * 8);
    }
  };

  const int nk = K / BK;
  issue(0, 0);
  for (int i = 0; i < nk; i++) {
    wait_async0();        // my async writes into buf (i&1) have landed
    __syncthreads();      // everyone's writes landed; prior reads retired
    if (i + 1 < nk) issue((i + 1) & 1, (i + 1) * BK);
    const __bf16* at = As[i & 1];
    const __bf16* bt = Bs[i & 1];
    v16bf af[4], bf_[4];
#pragma unroll
    for (int mi = 0; mi < 4; mi++) af[mi] = ld_frag(at, LDT, wm0 + mi * 16, 0, lane);
#pragma unroll
    for (int nj = 0; nj < 4; nj++) bf_[nj] = ld_frag(bt, LDT, wn0 + nj * 16, 0, lane);
#pragma unroll
    for (int mi = 0; mi < 4; mi++)
#pragma unroll
      for (int nj = 0; nj < 4; nj++)
        acc[mi][nj] = wmma_bf16f32(af[mi], bf_[nj], acc[mi][nj]);
  }

  // Epilogue. C/D layout: lane (g, n=lane&15), VGPR rr -> row rr + 8g.
  const int g = lane >> 4, cN = lane & 15;
#pragma unroll
  for (int mi = 0; mi < 4; mi++) {
#pragma unroll
    for (int nj = 0; nj < 4; nj++) {
      const int col = n0 + wn0 + nj * 16 + cN;
      const float bb = bias ? bias[col] : 0.f;
      AccU u; u.v = acc[mi][nj];
#pragma unroll
      for (int rr = 0; rr < 8; rr++) {
        const int row = m0 + wm0 + mi * 16 + rr + 8 * g;
        if (row >= Meff) continue;
        float v = u.f[rr] + bb;
        if constexpr (RELU) v = v > 0.f ? v : 0.f;
        if constexpr (OMODE == 0) {
          out_bf16[(long)row * N + col] = (__bf16)v;
        } else if constexpr (OMODE == 1) {
          out_f32[(long)row * N + col] = v;
        } else {
          const int orow = row_map[row];
          scat[(long)orow * ldscat + col] += scale[row] * v;  // unique rows
        }
      }
    }
  }
}

// ---------------------------------------------------------------------------
// Flash-style attention: one WG per (b, h, 64 query rows). 4 waves, each owns
// 16 q-rows. K/V blocks of 64 staged in LDS; QK^T and P*V via WMMA bf16.
// ---------------------------------------------------------------------------
__global__ __launch_bounds__(128) void attn_wmma(const __bf16* __restrict__ qkv,
                                                 __bf16* __restrict__ ctx) {
  constexpr int LDA = 72;  // 144B rows: 16B aligned, bank-friendly
  __shared__ __bf16 Qs[64 * LDA];
  __shared__ __bf16 Ks[64 * LDA];
  __shared__ __bf16 Vt[64 * LDA];  // transposed: [dh][key]
  __shared__ __bf16 Ps[64 * LDA];

  const int tid = threadIdx.x, lane = tid & 31, wv = tid >> 5;
  const int bh = blockIdx.y, b = bh >> 4, h = bh & 15;
  const int q0 = blockIdx.x * 64;
  const long rowstride = 3L * DD;
  const long baseQ = (long)(b * SS) * rowstride + h * DHH;
  const long baseK = baseQ + DD;
  const long baseV = baseQ + 2 * DD;
  const float scale = 0.125f;  // 1/sqrt(64)

  // Load Q block 64x64 via async LDS copies (8-elem chunks, 4/thread).
#pragma unroll
  for (int it = 0; it < 4; it++) {
    const int c = tid + it * 128;
    const int row = c >> 3, cc = c & 7;
    const uint32_t lq = (uint32_t)(uintptr_t)&Qs[row * LDA + cc * 8];
    async_cp16(lq, &qkv[baseQ + (long)(q0 + row) * rowstride + cc * 8]);
  }

  float m_i[8], l_i[8];
  v8f o[4];
#pragma unroll
  for (int r = 0; r < 8; r++) { m_i[r] = -3.0e38f; l_i[r] = 0.f; }
#pragma unroll
  for (int n = 0; n < 4; n++) o[n] = vzero8f();

  const int g = lane >> 4, cN = lane & 15;

  for (int kb = 0; kb < SS; kb += 64) {
    __syncthreads();  // prior Ks/Vt reads retired before overwrite
    // K block direct async copy; V block transposed through VGPRs.
#pragma unroll
    for (int it = 0; it < 4; it++) {
      const int c = tid + it * 128;
      const int row = c >> 3, cc = c & 7;
      const uint32_t lk = (uint32_t)(uintptr_t)&Ks[row * LDA + cc * 8];
      async_cp16(lk, &qkv[baseK + (long)(kb + row) * rowstride + cc * 8]);
      v8bf vv = *(const v8bf*)&qkv[baseV + (long)(kb + row) * rowstride + cc * 8];
#pragma unroll
      for (int u = 0; u < 8; u++) Vt[(cc * 8 + u) * LDA + row] = vv[u];
    }
    wait_async0();
    __syncthreads();

    // S = Q * K^T for this wave's 16 q-rows x 64 keys.
    v8f sa[4];
#pragma unroll
    for (int n = 0; n < 4; n++) sa[n] = vzero8f();
#pragma unroll
    for (int ks = 0; ks < 2; ks++) {
      v16bf aq = ld_frag(Qs, LDA, wv * 16, ks * 32, lane);
#pragma unroll
      for (int n = 0; n < 4; n++) {
        v16bf bk = ld_frag(Ks, LDA, n * 16, ks * 32, lane);
        sa[n] = wmma_bf16f32(aq, bk, sa[n]);
      }
    }

    // Online softmax. Row rr+8g lives in VGPR rr across the 16 lanes of
    // this half-wave -> shfl_xor masks 1..8 reduce within the half.
    AccU su[4];
#pragma unroll
    for (int n = 0; n < 4; n++) su[n].v = sa[n];
#pragma unroll
    for (int rr = 0; rr < 8; rr++) {
      float mx = su[0].f[rr];
#pragma unroll
      for (int n = 1; n < 4; n++) mx = fmaxf(mx, su[n].f[rr]);
#pragma unroll
      for (int off = 1; off < 16; off <<= 1) mx = fmaxf(mx, __shfl_xor(mx, off, 32));
      const float mblk = mx * scale;
      const float mnew = fmaxf(m_i[rr], mblk);
      const float corr = __expf(m_i[rr] - mnew);
      float ls = 0.f;
#pragma unroll
      for (int n = 0; n < 4; n++) {
        const float p = __expf(su[n].f[rr] * scale - mnew);
        su[n].f[rr] = p;
        ls += p;
      }
#pragma unroll
      for (int off = 1; off < 16; off <<= 1) ls += __shfl_xor(ls, off, 32);
      l_i[rr] = l_i[rr] * corr + ls;
      m_i[rr] = mnew;
#pragma unroll
      for (int n = 0; n < 4; n++) o[n][rr] *= corr;
    }

    // C-layout -> A-layout shuffle through LDS (per-wave rows, no barrier).
#pragma unroll
    for (int n = 0; n < 4; n++)
#pragma unroll
      for (int rr = 0; rr < 8; rr++)
        Ps[(wv * 16 + rr + 8 * g) * LDA + n * 16 + cN] = (__bf16)su[n].f[rr];

    // O += P * V
#pragma unroll
    for (int ks = 0; ks < 2; ks++) {
      v16bf ap = ld_frag(Ps, LDA, wv * 16, ks * 32, lane);
#pragma unroll
      for (int n = 0; n < 4; n++) {
        v16bf bv = ld_frag(Vt, LDA, n * 16, ks * 32, lane);
        o[n] = wmma_bf16f32(ap, bv, o[n]);
      }
    }
  }

#pragma unroll
  for (int rr = 0; rr < 8; rr++) {
    const float inv = 1.f / l_i[rr];
    const long tok = (long)b * SS + q0 + wv * 16 + rr + 8 * g;
#pragma unroll
    for (int n = 0; n < 4; n++)
      ctx[tok * DD + h * DHH + n * 16 + cN] = (__bf16)(o[n][rr] * inv);
  }
}

// ---------------------------------------------------------------------------
// Residual add + LayerNorm over D=1024. One token per block (256 thr).
// ---------------------------------------------------------------------------
__global__ __launch_bounds__(256) void add_ln(
    const float* __restrict__ a, const float* __restrict__ bsrc,
    const float* __restrict__ w, const float* __restrict__ bia,
    float* __restrict__ of32, __bf16* __restrict__ obf16) {
  const long t = blockIdx.x;
  const int tid = threadIdx.x;
  float4 xv = ((const float4*)(a + t * DD))[tid];
  if (bsrc) {
    const float4 bv = ((const float4*)(bsrc + t * DD))[tid];
    xv.x += bv.x; xv.y += bv.y; xv.z += bv.z; xv.w += bv.w;
  }
  float s = xv.x + xv.y + xv.z + xv.w;
  float q = xv.x * xv.x + xv.y * xv.y + xv.z * xv.z + xv.w * xv.w;
#pragma unroll
  for (int off = 16; off; off >>= 1) {
    s += __shfl_xor(s, off, 32);
    q += __shfl_xor(q, off, 32);
  }
  __shared__ float ssum[8], ssq[8];
  if ((tid & 31) == 0) { ssum[tid >> 5] = s; ssq[tid >> 5] = q; }
  __syncthreads();
  float ts = 0.f, tq = 0.f;
#pragma unroll
  for (int i = 0; i < 8; i++) { ts += ssum[i]; tq += ssq[i]; }
  const float mu = ts * (1.f / DD);
  const float var = tq * (1.f / DD) - mu * mu;
  const float inv = rsqrtf(var + 1e-5f);
  const float4 wv = ((const float4*)w)[tid];
  const float4 bv = ((const float4*)bia)[tid];
  const float y0 = (xv.x - mu) * inv * wv.x + bv.x;
  const float y1 = (xv.y - mu) * inv * wv.y + bv.y;
  const float y2 = (xv.z - mu) * inv * wv.z + bv.z;
  const float y3 = (xv.w - mu) * inv * wv.w + bv.w;
  if (of32) {
    float4 yy; yy.x = y0; yy.y = y1; yy.z = y2; yy.w = y3;
    ((float4*)(of32 + t * DD))[tid] = yy;
  }
  if (obf16) {
    __bf16* op = obf16 + t * DD + tid * 4;
    op[0] = (__bf16)y0; op[1] = (__bf16)y1; op[2] = (__bf16)y2; op[3] = (__bf16)y3;
  }
}

// ---------------------------------------------------------------------------
// Router: softmax over E=8 logits, top-2 renormalized, per-expert token
// lists via int atomics (order-independent), deterministic imp partials.
// ---------------------------------------------------------------------------
__global__ __launch_bounds__(256) void router_kernel(
    const float* __restrict__ x, const float* __restrict__ rw,
    const float* __restrict__ rb, int* __restrict__ counts,
    int* __restrict__ lists, float* __restrict__ wlists,
    float* __restrict__ partials) {
  const int wv = threadIdx.x >> 5, lane = threadIdx.x & 31;
  const int t = blockIdx.x * 8 + wv;
  float acc[EE];
#pragma unroll
  for (int e = 0; e < EE; e++) acc[e] = 0.f;
  for (int i = lane; i < DD; i += 32) {
    const float xv = x[(long)t * DD + i];
#pragma unroll
    for (int e = 0; e < EE; e++) acc[e] += xv * rw[e * DD + i];
  }
#pragma unroll
  for (int off = 16; off; off >>= 1)
#pragma unroll
    for (int e = 0; e < EE; e++) acc[e] += __shfl_xor(acc[e], off, 32);

  __shared__ float probs_s[8][EE];
  if (lane == 0) {
    float p[EE];
    float mx = -3.0e38f;
#pragma unroll
    for (int e = 0; e < EE; e++) { p[e] = acc[e] + rb[e]; mx = fmaxf(mx, p[e]); }
    float sum = 0.f;
#pragma unroll
    for (int e = 0; e < EE; e++) { p[e] = __expf(p[e] - mx); sum += p[e]; }
    const float isum = 1.f / sum;
#pragma unroll
    for (int e = 0; e < EE; e++) p[e] *= isum;
    int i1 = 0;
#pragma unroll
    for (int e = 1; e < EE; e++) if (p[e] > p[i1]) i1 = e;
    int i2 = (i1 == 0) ? 1 : 0;
#pragma unroll
    for (int e = 0; e < EE; e++) if (e != i1 && p[e] > p[i2]) i2 = e;
    const float rs = 1.f / (p[i1] + p[i2]);
    const int s1 = atomicAdd(&counts[i1], 1);
    lists[i1 * TT + s1] = t; wlists[i1 * TT + s1] = p[i1] * rs;
    const int s2 = atomicAdd(&counts[i2], 1);
    lists[i2 * TT + s2] = t; wlists[i2 * TT + s2] = p[i2] * rs;
#pragma unroll
    for (int e = 0; e < EE; e++) probs_s[wv][e] = p[e];
  }
  __syncthreads();
  if (threadIdx.x < EE) {
    float s = 0.f;
#pragma unroll
    for (int ww = 0; ww < 8; ww++) s += probs_s[ww][threadIdx.x];
    partials[blockIdx.x * EE + threadIdx.x] = s;
  }
}

__global__ void aux_kernel(const float* __restrict__ partials, int nblk,
                           const int* __restrict__ counts,
                           float* __restrict__ out_aux) {
  __shared__ float s[EE];
  if (threadIdx.x < EE) {
    float a = 0.f;
    for (int i = 0; i < nblk; i++) a += partials[i * EE + threadIdx.x];
    s[threadIdx.x] = a;
  }
  __syncthreads();
  if (threadIdx.x == 0) {
    float aux = 0.f;
    const float invT = 1.f / (float)TT;
#pragma unroll
    for (int e = 0; e < EE; e++)
      aux += ((float)counts[e] * invT) * (s[e] * invT);
    *out_aux = (float)EE * aux;
  }
}

__global__ void zero_kernel(float* __restrict__ p, long n, int* __restrict__ counts) {
  const long i = (long)blockIdx.x * blockDim.x + threadIdx.x;
  if (i < n) p[i] = 0.f;
  if (i < EE) counts[i] = 0;
}

// f32 -> bf16 conversion (vectorized), n must be a multiple of 4.
__global__ void cvt_bf16(const float4* __restrict__ in, __bf16* __restrict__ out,
                         long n4) {
  const long i = (long)blockIdx.x * blockDim.x + threadIdx.x;
  if (i >= n4) return;
  const float4 f = in[i];
  __bf16* op = out + i * 4;
  op[0] = (__bf16)f.x; op[1] = (__bf16)f.y; op[2] = (__bf16)f.z; op[3] = (__bf16)f.w;
}

// ---------------------------------------------------------------------------
extern "C" void kernel_launch(void* const* d_in, const int* in_sizes, int n_in,
                              void* d_out, int out_size, void* d_ws,
                              size_t ws_size, hipStream_t stream) {
  (void)in_sizes; (void)n_in; (void)out_size; (void)ws_size;
  const float* src        = (const float*)d_in[0];
  const float* in_proj_w  = (const float*)d_in[1];
  const float* in_proj_b  = (const float*)d_in[2];
  const float* out_proj_w = (const float*)d_in[3];
  const float* out_proj_b = (const float*)d_in[4];
  const float* norm1_w    = (const float*)d_in[5];
  const float* norm1_b    = (const float*)d_in[6];
  const float* norm2_w    = (const float*)d_in[7];
  const float* norm2_b    = (const float*)d_in[8];
  const float* router_w   = (const float*)d_in[9];
  const float* router_b   = (const float*)d_in[10];
  const float* w1         = (const float*)d_in[11];
  const float* b1         = (const float*)d_in[12];
  const float* w2         = (const float*)d_in[13];
  const float* b2         = (const float*)d_in[14];
  float* out = (float*)d_out;

  // Workspace layout (~404 MB), 256B aligned segments.
  char* ws = (char*)d_ws;
  size_t off = 0;
  auto alloc = [&](size_t bytes) {
    void* p = ws + off;
    off += (bytes + 255) & ~(size_t)255;
    return p;
  };
  __bf16* src_b   = (__bf16*)alloc((size_t)TT * DD * 2);
  __bf16* wq_b    = (__bf16*)alloc((size_t)3 * DD * DD * 2);
  __bf16* wo_b    = (__bf16*)alloc((size_t)DD * DD * 2);
  __bf16* w1_b    = (__bf16*)alloc((size_t)EE * HIDD * DD * 2);
  __bf16* w2_b    = (__bf16*)alloc((size_t)EE * DD * HIDD * 2);
  __bf16* qkv_b   = (__bf16*)alloc((size_t)TT * 3 * DD * 2);
  __bf16* ctx_b   = (__bf16*)alloc((size_t)TT * DD * 2);
  float*  attnout = (float*) alloc((size_t)TT * DD * 4);
  float*  x_f32   = (float*) alloc((size_t)TT * DD * 4);
  __bf16* x_b     = (__bf16*)alloc((size_t)TT * DD * 2);
  float*  moe_f32 = (float*) alloc((size_t)TT * DD * 4);
  __bf16* h_b     = (__bf16*)alloc((size_t)TT * HIDD * 2);
  int*    lists   = (int*)   alloc((size_t)EE * TT * 4);
  float*  wlists  = (float*) alloc((size_t)EE * TT * 4);
  int*    counts  = (int*)   alloc(EE * 4);
  float*  partials= (float*) alloc((size_t)(TT / 8) * EE * 4);

  const long nz = (long)TT * DD;
  zero_kernel<<<dim3((unsigned)((nz + 255) / 256)), 256, 0, stream>>>(moe_f32, nz, counts);

  // bf16 conversions (weights + src): memory-bound, ~10us total at 23 TB/s.
  auto cvt = [&](const float* in, __bf16* o, long n) {
    const long n4 = n / 4;
    cvt_bf16<<<dim3((unsigned)((n4 + 255) / 256)), 256, 0, stream>>>(
        (const float4*)in, o, n4);
  };
  cvt(src, src_b, (long)TT * DD);
  cvt(in_proj_w, wq_b, (long)3 * DD * DD);
  cvt(out_proj_w, wo_b, (long)DD * DD);
  cvt(w1, w1_b, (long)EE * HIDD * DD);
  cvt(w2, w2_b, (long)EE * DD * HIDD);

  // QKV projection: [T,D] x [3D,D]^T -> bf16 [T,3D]
  gemm_wmma<0, false, false><<<dim3(3 * DD / 128, TT / 128), 128, 0, stream>>>(
      src_b, wq_b, in_proj_b, TT, 3 * DD, DD, qkv_b, nullptr, nullptr,
      nullptr, nullptr, nullptr, 0);

  // Attention -> ctx bf16 [T,D]
  attn_wmma<<<dim3(SS / 64, BB * HHH), 128, 0, stream>>>(qkv_b, ctx_b);

  // Out projection: [T,D] x [D,D]^T -> f32 [T,D]
  gemm_wmma<1, false, false><<<dim3(DD / 128, TT / 128), 128, 0, stream>>>(
      ctx_b, wo_b, out_proj_b, TT, DD, DD, nullptr, attnout, nullptr,
      nullptr, nullptr, nullptr, 0);

  // x = LN(src + attn_out) -> f32 + bf16 copies
  add_ln<<<TT, 256, 0, stream>>>(src, attnout, norm1_w, norm1_b, x_f32, x_b);

  // Router + aux-loss partials
  router_kernel<<<TT / 8, 256, 0, stream>>>(x_f32, router_w, router_b, counts,
                                            lists, wlists, partials);
  aux_kernel<<<1, 32, 0, stream>>>(partials, TT / 8, counts, out + (long)TT * DD);

  // Sparse MoE: per expert, gathered FFN1 (relu) then scaled scatter-add FFN2.
  for (int e = 0; e < EE; e++) {
    gemm_wmma<0, true, true><<<dim3(HIDD / 128, TT / 128), 128, 0, stream>>>(
        x_b, w1_b + (size_t)e * HIDD * DD, b1 + e * HIDD, TT, HIDD, DD, h_b,
        nullptr, lists + e * TT, counts + e, nullptr, nullptr, 0);
    gemm_wmma<2, false, false><<<dim3(DD / 128, TT / 128), 128, 0, stream>>>(
        h_b, w2_b + (size_t)e * DD * HIDD, b2 + e * DD, TT, DD, HIDD, nullptr,
        nullptr, lists + e * TT, counts + e, wlists + e * TT, moe_f32, DD);
  }

  // out = LN(x + moe_out)
  add_ln<<<TT, 256, 0, stream>>>(x_f32, moe_f32, norm2_w, norm2_b, out, nullptr);
}
